// MultiHeadMomentumSMoELayer_9320079032506
// MI455X (gfx1250) — matrix operand order repaired and checked by reading, not compiled, not run on hardware
//
#include <hip/hip_runtime.h>
#include <hip/hip_bf16.h>
#include <math.h>

// ---------------- problem constants ----------------
#define BB 4
#define SS 2048
#define DD 1024
#define HH 8
#define HD 128
#define FF 512
#define EE 8
#define TT (BB * SS * HH)      // 65536 tokens
#define MM (BB * SS)           // 8192 rows for D-wide GEMMs
#define MU 0.7f
#define LN_EPS 1e-5f

// ---------------- WMMA types ----------------
typedef __attribute__((ext_vector_type(16))) __bf16 v16bf;
typedef __attribute__((ext_vector_type(8)))  float  v8f;

struct __attribute__((aligned(16))) U128 { unsigned int x, y, z, w; };
struct U256 { U128 lo, hi; };

__device__ inline unsigned short f2bf(float f) {
    union { float f; unsigned u; } v; v.f = f;
    unsigned u = v.u;
    u += 0x7FFFu + ((u >> 16) & 1u);     // round-to-nearest-even
    return (unsigned short)(u >> 16);
}

__device__ inline v8f wmma_bf16(v16bf a, v16bf b, v8f c) {
    return __builtin_amdgcn_wmma_f32_16x16x32_bf16(
        false, a, false, b, (short)0, c, false, false);
}

// A fragment 16x32 (MxK) from LDS, row-major [M][K], row stride ldk (elems).
// Lane l<16: row M=l, K = 0..7 (elems 0-7) and 16..23 (elems 8-15).
// Lane l>=16: row M=l-16, K = 8..15 and 24..31.   (ISA 7.12.2)
__device__ inline v16bf load_a(const unsigned short* p, int ldk) {
    int lane = threadIdx.x & 31;
    const unsigned short* r = p + (lane & 15) * ldk + ((lane >> 4) << 3);
    U256 pk;
    pk.lo = *(const U128*)r;
    pk.hi = *(const U128*)(r + 16);
    return __builtin_bit_cast(v16bf, pk);
}

// B fragment 32x16 (KxN) from LDS stored TRANSPOSED [N][K], row stride ldk.
// Lane l<16: N=l, K = 0..7 and 16..23; lane l>=16: N=l-16, K = 8..15 and 24..31.
__device__ inline v16bf load_bt(const unsigned short* p, int ldk) {
    int lane = threadIdx.x & 31;
    const unsigned short* r = p + (lane & 15) * ldk + ((lane >> 4) << 3);
    U256 pk;
    pk.lo = *(const U128*)r;
    pk.hi = *(const U128*)(r + 16);
    return __builtin_bit_cast(v16bf, pk);
}

// ---------------- helper kernels ----------------
__global__ void cvt_f32_bf16(const float* __restrict__ src,
                             unsigned short* __restrict__ dst, int n4) {
    for (int i = blockIdx.x * blockDim.x + threadIdx.x; i < n4;
         i += gridDim.x * blockDim.x) {
        float4 v = ((const float4*)src)[i];
        unsigned long long o =
            (unsigned long long)f2bf(v.x)        |
            ((unsigned long long)f2bf(v.y) << 16)|
            ((unsigned long long)f2bf(v.z) << 32)|
            ((unsigned long long)f2bf(v.w) << 48);
        *(unsigned long long*)(dst + 4ull * i) = o;
    }
}

// transpose + convert: dst[c][r] = bf16(src[r][c]); src is R x C, batched on z.
__global__ void cvt_transpose(const float* __restrict__ src,
                              unsigned short* __restrict__ dst, int R, int C) {
    __shared__ float tile[32][33];
    const size_t boff = (size_t)blockIdx.z * R * C;
    const int r0 = blockIdx.y * 32, c0 = blockIdx.x * 32;
    const int tx = threadIdx.x, ty = threadIdx.y;   // 32 x 8
#pragma unroll
    for (int i = 0; i < 32; i += 8)
        tile[ty + i][tx] = src[boff + (size_t)(r0 + ty + i) * C + c0 + tx];
    __syncthreads();
#pragma unroll
    for (int i = 0; i < 32; i += 8)
        dst[boff + (size_t)(c0 + ty + i) * R + r0 + tx] = f2bf(tile[tx][ty + i]);
}

__global__ void zero_ws(float* __restrict__ eout, int* __restrict__ cnt, int n4) {
    int gid = blockIdx.x * blockDim.x + threadIdx.x;
    if (gid < EE) cnt[gid] = 0;
    float4 z = make_float4(0.f, 0.f, 0.f, 0.f);
    for (int i = gid; i < n4; i += gridDim.x * blockDim.x)
        ((float4*)eout)[i] = z;
}

// ---------------- 128x128-tile bf16 GEMM: C = A*Bt^T + bias ----------------
// A [M,K] bf16 row-major, Bt [N,K] bf16 row-major (pre-transposed weights).
// 256 threads = 8 waves; each wave owns a 32x64 sub-tile (2x4 f32 accum
// tiles): 2 A-frags + 4 B-frags per 8 WMMAs, each B frag feeds 2 WMMAs.
// Double-buffered LDS (K-chunk 64), rotate-preloaded B fragments so LDS
// reads overlap the matrix pipe instead of s_wait_dscnt 0 per WMMA.
__global__ __launch_bounds__(256)
void gemm_bf16(const unsigned short* __restrict__ A,
               const unsigned short* __restrict__ Bt,
               const float* __restrict__ bias,
               float* __restrict__ Cf, unsigned short* __restrict__ Cbf,
               int M, int N, int K) {
    __shared__ __align__(16) unsigned short As[2][128 * 64];
    __shared__ __align__(16) unsigned short Bs[2][128 * 64];
    const int tid  = threadIdx.x;
    const int wave = tid >> 5;
    const int lane = tid & 31;
    const int m0 = blockIdx.y * 128;
    const int n0 = blockIdx.x * 128;
    const int wm = (wave >> 1) * 32;      // wave row group: 0/32/64/96
    const int wn = (wave & 1) * 64;       // wave col group: 0/64

    v8f acc[2][4];
#pragma unroll
    for (int i = 0; i < 2; ++i)
#pragma unroll
        for (int j = 0; j < 4; ++j) acc[i][j] = v8f{};

    auto stage = [&](int k0, int buf) {
#pragma unroll
        for (int i = tid; i < 1024; i += 256) {
            int r = i >> 3, c8 = i & 7;
            *(U128*)&As[buf][r * 64 + c8 * 8] =
                *(const U128*)&A[(size_t)(m0 + r) * K + k0 + c8 * 8];
        }
#pragma unroll
        for (int i = tid; i < 1024; i += 256) {
            int r = i >> 3, c8 = i & 7;
            *(U128*)&Bs[buf][r * 64 + c8 * 8] =
                *(const U128*)&Bt[(size_t)(n0 + r) * K + k0 + c8 * 8];
        }
    };

    const int nk = K / 64;
    stage(0, 0);
    for (int it = 0; it < nk; ++it) {
        __syncthreads();
        if (it + 1 < nk) stage((it + 1) * 64, (it + 1) & 1);
        const int buf = it & 1;
#pragma unroll
        for (int ks = 0; ks < 2; ++ks) {
            v16bf a0 = load_a(&As[buf][(wm +  0) * 64 + ks * 32], 64);
            v16bf a1 = load_a(&As[buf][(wm + 16) * 64 + ks * 32], 64);
            v16bf bc = load_bt(&Bs[buf][(wn) * 64 + ks * 32], 64);
#pragma unroll
            for (int j = 0; j < 4; ++j) {
                v16bf bn = bc;
                if (j < 3)
                    bn = load_bt(&Bs[buf][(wn + (j + 1) * 16) * 64 + ks * 32], 64);
                acc[0][j] = wmma_bf16(a0, bc, acc[0][j]);
                acc[1][j] = wmma_bf16(a1, bc, acc[1][j]);
                bc = bn;
            }
        }
    }
    // epilogue
    const int halfm = (lane >> 4) << 3;
    const int col0  = lane & 15;
#pragma unroll
    for (int i = 0; i < 2; ++i)
#pragma unroll
        for (int j = 0; j < 4; ++j) {
            int colg = n0 + wn + j * 16 + col0;
            float bv = bias[colg];
#pragma unroll
            for (int r = 0; r < 8; ++r) {
                int rowg = m0 + wm + i * 16 + halfm + r;
                float v = acc[i][j][r] + bv;
                if (Cf)  Cf[(size_t)rowg * N + colg] = v;
                if (Cbf) Cbf[(size_t)rowg * N + colg] = f2bf(v);
            }
        }
}

// ---------------- gate + top-2 routing ----------------
__global__ void gate_topk(const float* __restrict__ t,
                          const float* __restrict__ gW,
                          const float* __restrict__ gb,
                          float* __restrict__ gates2,
                          int* __restrict__ elist, int* __restrict__ cnt) {
    int tok = blockIdx.x * blockDim.x + threadIdx.x;
    if (tok >= TT) return;
    float lg[EE];
#pragma unroll
    for (int e = 0; e < EE; ++e) lg[e] = gb[e];
    const float4* tr = (const float4*)(t + (size_t)tok * HD);
#pragma unroll 4
    for (int d4 = 0; d4 < HD / 4; ++d4) {
        float4 tv = tr[d4];
#pragma unroll
        for (int c = 0; c < 4; ++c) {
            float x = (&tv.x)[c];
            const float* gr = gW + (size_t)(d4 * 4 + c) * EE;
#pragma unroll
            for (int e = 0; e < EE; ++e) lg[e] += x * gr[e];
        }
    }
    int i0 = 0; float v0 = lg[0];
#pragma unroll
    for (int e = 1; e < EE; ++e) if (lg[e] > v0) { v0 = lg[e]; i0 = e; }
    int i1 = -1; float v1 = -3.4e38f;
#pragma unroll
    for (int e = 0; e < EE; ++e)
        if (e != i0 && lg[e] > v1) { v1 = lg[e]; i1 = e; }
    float ex = __expf(v1 - v0);
    float g0 = 1.0f / (1.0f + ex);
    float g1 = ex * g0;
    gates2[tok * 2 + 0] = g0;
    gates2[tok * 2 + 1] = g1;
    int p0 = atomicAdd(&cnt[i0], 1);
    elist[i0 * TT + p0] = tok * 2;
    int p1 = atomicAdd(&cnt[i1], 1);
    elist[i1 * TT + p1] = tok * 2 + 1;
}

// ---------------- routed fused expert FFN ----------------
// grid (tiles=TT/128, E). Processes <=128 routed rows of expert e:
// h = relu(t@W1[e]+b1)  (FF chunked by 32), eo += h@W2[e]; scatter g*(eo+b2).
// W1t = W1^T per expert [FF][HD]; W2t = W2^T per expert [HD][FF].
__global__ __launch_bounds__(256)
void expert_ffn(const unsigned short* __restrict__ t_bf,
                const unsigned short* __restrict__ W1t,
                const unsigned short* __restrict__ W2t,
                const float* __restrict__ b1, const float* __restrict__ b2,
                const int* __restrict__ elist, const int* __restrict__ cnt,
                const float* __restrict__ gates2,
                float* __restrict__ expert_out) {
    const int e = blockIdx.y;
    const int n = cnt[e];
    const int m0 = blockIdx.x * 128;
    if (m0 >= n) return;

    __shared__ __align__(16) unsigned short As[128 * 128];    // gathered tokens
    __shared__ __align__(16) unsigned short B1s[32 * 128];    // W1t chunk [f][d]
    __shared__ __align__(16) unsigned short Hs[128 * 32];     // relu(h) bf16 [tok][f]
    __shared__ __align__(16) unsigned short B2s[128 * 32];    // W2t chunk [d][f]

    const int tid  = threadIdx.x;
    const int wave = tid >> 5;
    const int lane = tid & 31;
    const int wm = (wave >> 1) * 32;      // eo-phase: wave 32x64 sub-tile
    const int wn = (wave & 1) * 64;

    // gather A rows (zero-pad tail)
#pragma unroll
    for (int i = tid; i < 2048; i += 256) {
        int r = i >> 4, c4 = i & 15;
        int gidx = m0 + r;
        U128 v;
        if (gidx < n) {
            int tok = elist[e * TT + gidx] >> 1;
            v = *(const U128*)&t_bf[(size_t)tok * HD + c4 * 8];
        } else {
            v = U128{0u, 0u, 0u, 0u};
        }
        *(U128*)&As[r * 128 + c4 * 8] = v;
    }

    v8f acc[2][4];
#pragma unroll
    for (int i = 0; i < 2; ++i)
#pragma unroll
        for (int j = 0; j < 4; ++j) acc[i][j] = v8f{};

    const int halfm = (lane >> 4) << 3;
    const int col0  = lane & 15;

    __syncthreads();
    for (int f0 = 0; f0 < FF; f0 += 32) {
        // stage W1t[e][f0:f0+32][0:128] and W2t[e][0:128][f0:f0+32]
#pragma unroll
        for (int i = tid; i < 512; i += 256) {
            int r = i >> 4, c4 = i & 15;
            *(U128*)&B1s[r * 128 + c4 * 8] =
                *(const U128*)&W1t[(size_t)e * FF * HD + (size_t)(f0 + r) * HD + c4 * 8];
        }
#pragma unroll
        for (int i = tid; i < 512; i += 256) {
            int r = i >> 2, c4 = i & 3;
            *(U128*)&B2s[r * 32 + c4 * 8] =
                *(const U128*)&W2t[(size_t)e * HD * FF + (size_t)r * FF + f0 + c4 * 8];
        }
        __syncthreads();

        // h chunk [128,32]: each wave a 16x32 strip (2 tiles), K=128 (4 steps),
        // rotate-preloaded fragments.
        v8f h0 = v8f{}, h1 = v8f{};
        v16bf a_c  = load_a(&As[wave * 16 * 128 + 0], 128);
        v16bf b0_c = load_bt(&B1s[0 * 16 * 128 + 0], 128);
        v16bf b1_c = load_bt(&B1s[1 * 16 * 128 + 0], 128);
#pragma unroll
        for (int ks = 0; ks < 4; ++ks) {
            v16bf a_n = a_c, b0_n = b0_c, b1_n = b1_c;
            if (ks < 3) {
                a_n  = load_a(&As[wave * 16 * 128 + (ks + 1) * 32], 128);
                b0_n = load_bt(&B1s[0 * 16 * 128 + (ks + 1) * 32], 128);
                b1_n = load_bt(&B1s[1 * 16 * 128 + (ks + 1) * 32], 128);
            }
            h0 = wmma_bf16(a_c, b0_c, h0);
            h1 = wmma_bf16(a_c, b1_c, h1);
            a_c = a_n; b0_c = b0_n; b1_c = b1_n;
        }
        // bias + relu + bf16 into Hs
        float bva = b1[e * FF + f0 + col0];
        float bvb = b1[e * FF + f0 + 16 + col0];
#pragma unroll
        for (int r = 0; r < 8; ++r) {
            int row = wave * 16 + halfm + r;
            float va = h0[r] + bva; va = va > 0.f ? va : 0.f;
            float vb = h1[r] + bvb; vb = vb > 0.f ? vb : 0.f;
            Hs[row * 32 + col0]      = f2bf(va);
            Hs[row * 32 + 16 + col0] = f2bf(vb);
        }
        __syncthreads();

        // eo += Hs(32x32 per wave-rowgroup) @ W2chunk(32x64 per wave-colgroup)
        v16bf a0 = load_a(&Hs[(wm +  0) * 32], 32);
        v16bf a1 = load_a(&Hs[(wm + 16) * 32], 32);
        v16bf bc = load_bt(&B2s[wn * 32], 32);
#pragma unroll
        for (int j = 0; j < 4; ++j) {
            v16bf bn = bc;
            if (j < 3) bn = load_bt(&B2s[(wn + (j + 1) * 16) * 32], 32);
            acc[0][j] = wmma_bf16(a0, bc, acc[0][j]);
            acc[1][j] = wmma_bf16(a1, bc, acc[1][j]);
            bc = bn;
        }
        __syncthreads();
    }

    // epilogue: g * (eo + b2) -> atomic scatter (exactly 2 adds per element,
    // zeroed buffer + commutative f32 add => deterministic)
    int   codes[2][8];
    float gg[2][8];
#pragma unroll
    for (int i = 0; i < 2; ++i)
#pragma unroll
        for (int r = 0; r < 8; ++r) {
            int gidx = m0 + wm + i * 16 + halfm + r;
            int code = (gidx < n) ? elist[e * TT + gidx] : -1;
            codes[i][r] = code;
            gg[i][r] = (code >= 0) ? gates2[code] : 0.0f;
        }
#pragma unroll
    for (int i = 0; i < 2; ++i)
#pragma unroll
        for (int j = 0; j < 4; ++j) {
            int col = wn + j * 16 + col0;
            float bc = b2[e * HD + col];
#pragma unroll
            for (int r = 0; r < 8; ++r) {
                if (codes[i][r] >= 0) {
                    int tok = codes[i][r] >> 1;
                    atomicAdd(&expert_out[(size_t)tok * HD + col],
                              gg[i][r] * (acc[i][j][r] + bc));
                }
            }
        }
}

// ---------------- momentum + layernorm (one wave per token) ----------------
__global__ __launch_bounds__(256)
void momentum_ln(const float* __restrict__ t,
                 const float* __restrict__ eout,
                 const float* __restrict__ mom_in,
                 const float* __restrict__ ln_g, const float* __restrict__ ln_b,
                 float* __restrict__ new_mom,
                 unsigned short* __restrict__ ln_bf) {
    const int wave = threadIdx.x >> 5;
    const int lane = threadIdx.x & 31;
    const int tok  = blockIdx.x * 8 + wave;
    const size_t base = (size_t)tok * HD + lane * 4;

    float4 eo = *(const float4*)(eout + base);
    float4 mi = *(const float4*)(mom_in + base);
    float4 tv = *(const float4*)(t + base);

    float o[4], nm[4];
#pragma unroll
    for (int c = 0; c < 4; ++c) {
        float ec = (&eo.x)[c], mc = (&mi.x)[c];
        nm[c] = -ec + MU * mc;
        o[c]  = (&tv.x)[c] + nm[c];          // GAMMA = 1
    }
    *(float4*)(new_mom + base) = make_float4(nm[0], nm[1], nm[2], nm[3]);

    float s = o[0] + o[1] + o[2] + o[3];
#pragma unroll
    for (int off = 16; off >= 1; off >>= 1) s += __shfl_xor(s, off, 32);
    float mean = s * (1.0f / HD);
    float ss = 0.f;
#pragma unroll
    for (int c = 0; c < 4; ++c) { float d = o[c] - mean; ss += d * d; }
#pragma unroll
    for (int off = 16; off >= 1; off >>= 1) ss += __shfl_xor(ss, off, 32);
    float inv = rsqrtf(ss * (1.0f / HD) + LN_EPS);
#pragma unroll
    for (int c = 0; c < 4; ++c) {
        int d = lane * 4 + c;
        float v = (o[c] - mean) * inv * ln_g[d] + ln_b[d];
        ln_bf[base + c] = f2bf(v);
    }
}

// ---------------- host launcher ----------------
extern "C" void kernel_launch(void* const* d_in, const int* in_sizes, int n_in,
                              void* d_out, int out_size, void* d_ws, size_t ws_size,
                              hipStream_t stream) {
    const float* x       = (const float*)d_in[0];
    const float* mom     = (const float*)d_in[1];
    const float* split_W = (const float*)d_in[2];
    const float* split_b = (const float*)d_in[3];
    const float* gate_W  = (const float*)d_in[4];
    const float* gate_b  = (const float*)d_in[5];
    const float* W1      = (const float*)d_in[6];
    const float* b1      = (const float*)d_in[7];
    const float* W2      = (const float*)d_in[8];
    const float* b2      = (const float*)d_in[9];
    const float* ln_g    = (const float*)d_in[10];
    const float* ln_b    = (const float*)d_in[11];
    const float* merge_W = (const float*)d_in[12];
    const float* merge_b = (const float*)d_in[13];

    float* out_final = (float*)d_out;                       // [B,S,D]
    float* out_mom   = (float*)d_out + (size_t)MM * DD;     // [T,HD]

    // workspace layout
    char* ws = (char*)d_ws;
    size_t off = 0;
    auto take = [&](size_t bytes) { char* p = ws + off; off = (off + bytes + 255) & ~(size_t)255; return p; };
    unsigned short* x_bf    = (unsigned short*)take((size_t)MM * DD * 2);
    unsigned short* sWt_bf  = (unsigned short*)take((size_t)DD * DD * 2);   // split_W^T
    unsigned short* mWt_bf  = (unsigned short*)take((size_t)DD * DD * 2);   // merge_W^T
    unsigned short* W1t_bf  = (unsigned short*)take((size_t)EE * HD * FF * 2); // per-e [FF][HD]
    unsigned short* W2t_bf  = (unsigned short*)take((size_t)EE * FF * HD * 2); // per-e [HD][FF]
    float*          t_f32   = (float*)take((size_t)MM * DD * 4);
    unsigned short* t_bf    = (unsigned short*)take((size_t)MM * DD * 2);
    unsigned short* ln_bf16 = (unsigned short*)take((size_t)MM * DD * 2);
    float*          gates2  = (float*)take((size_t)TT * 2 * 4);
    int*            elist   = (int*)take((size_t)EE * TT * 4);
    int*            cnt     = (int*)take(256);
    float*          eout    = (float*)take((size_t)TT * HD * 4);
    (void)ws_size; (void)in_sizes; (void)n_in; (void)out_size;

    // 0) zero expert accumulator + routing counters (harness replays the graph)
    zero_ws<<<2048, 256, 0, stream>>>(eout, cnt, (int)((size_t)TT * HD / 4));

    // 1) fp32 -> bf16 conversions; weights transposed so GEMM B-tiles stage
    //    into LDS [N][K] and fragments load as 2x contiguous b128 per lane.
    cvt_f32_bf16<<<2048, 256, 0, stream>>>(x, x_bf, MM * DD / 4);
    cvt_transpose<<<dim3(32, 32, 1), dim3(32, 8), 0, stream>>>(split_W, sWt_bf, DD, DD);
    cvt_transpose<<<dim3(32, 32, 1), dim3(32, 8), 0, stream>>>(merge_W, mWt_bf, DD, DD);
    cvt_transpose<<<dim3(FF / 32, HD / 32, EE), dim3(32, 8), 0, stream>>>(W1, W1t_bf, HD, FF);
    cvt_transpose<<<dim3(HD / 32, FF / 32, EE), dim3(32, 8), 0, stream>>>(W2, W2t_bf, FF, HD);

    // 2) split GEMM: t = x @ split_W + split_b  (fp32 + bf16 outputs)
    gemm_bf16<<<dim3(DD / 128, MM / 128), 256, 0, stream>>>(
        x_bf, sWt_bf, split_b, t_f32, t_bf, MM, DD, DD);

    // 3) top-2 gate + routing lists
    gate_topk<<<TT / 256, 256, 0, stream>>>(t_f32, gate_W, gate_b, gates2, elist, cnt);

    // 4) routed expert FFN (2/8 of the dense reference FLOPs, same math)
    expert_ffn<<<dim3(TT / 128, EE), 256, 0, stream>>>(
        t_bf, W1t_bf, W2t_bf, b1, b2, elist, cnt, gates2, eout);

    // 5) momentum + layernorm (writes new_momentum output + bf16 for merge)
    momentum_ln<<<TT / 8, 256, 0, stream>>>(
        t_f32, eout, mom, ln_g, ln_b, out_mom, ln_bf16);

    // 6) merge GEMM: final = ln @ merge_W + merge_b
    gemm_bf16<<<dim3(DD / 128, MM / 128), 256, 0, stream>>>(
        ln_bf16, mWt_bf, merge_b, out_final, (unsigned short*)nullptr, MM, DD, DD);
}
